// RGCNEncoder_36352603193997
// MI455X (gfx1250) — matrix operand order repaired and checked by reading, not compiled, not run on hardware
//
#include <hip/hip_runtime.h>
#include <hip/hip_bf16.h>

// ---------------------------------------------------------------------------
// RGCN encoder for MI455X (gfx1250, wave32).
// Strategy: bf16 WMMA (v_wmma_f32_16x16x32_bf16) for both the per-edge
// relation GEMMs and the root GEMM; edges bucketed by relation (16-aligned,
// padded) so each 16-edge tile shares a single B matrix; scatter-add into an
// L2-resident fp32 accumulator with native global_atomic_add_f32 (no-return,
// STOREcnt-tracked) — avoids the CAS-loop expansion of generic fp atomics.
// ---------------------------------------------------------------------------

typedef __bf16 bf16_t;
typedef __bf16 v16bf __attribute__((ext_vector_type(16)));
typedef float  v8f   __attribute__((ext_vector_type(8)));

#define D_DIM 256
#define R_REL 8
#define L_LAY 3

__device__ __forceinline__ bf16_t f32_to_bf16(float f) {
  unsigned u = __builtin_bit_cast(unsigned, f);
  unsigned r = u + 0x7FFFu + ((u >> 16) & 1u);   // round-to-nearest-even
  unsigned short h = (unsigned short)(r >> 16);
  return __builtin_bit_cast(bf16_t, h);
}

// Native fp32 scatter-add (L2 atomic unit), bypassing atomicrmw CAS expansion.
__device__ __forceinline__ void atomic_add_f32(float* p, float v) {
  asm volatile("global_atomic_add_f32 %0, %1, off" :: "v"(p), "v"(v) : "memory");
}

// ------------------------- bucketing kernels -------------------------------

__global__ void k_zero_counts(int* cnt) {
  if (threadIdx.x < R_REL) cnt[threadIdx.x] = 0;
}

__global__ void k_count(const int* __restrict__ etype, int* __restrict__ cnt, int E) {
  int e = blockIdx.x * 256 + threadIdx.x;
  if (e < E) atomicAdd(&cnt[etype[e]], 1);
}

// 16-aligned per-relation region starts so every 16-edge tile is one relation.
__global__ void k_scan(const int* __restrict__ cnt, int* __restrict__ wcur) {
  if (threadIdx.x == 0) {
    int cur = 0;
    for (int r = 0; r < R_REL; ++r) {
      wcur[r] = cur;
      cur = (cur + cnt[r] + 15) & ~15;
    }
  }
}

__global__ void k_fill(int* __restrict__ slot_src, int slots) {
  int i = blockIdx.x * 256 + threadIdx.x;
  if (i < slots) slot_src[i] = -1;               // pad sentinel
}

__global__ void k_scatter(const int* __restrict__ esrc, const int* __restrict__ edst,
                          const int* __restrict__ etype, int* __restrict__ wcur,
                          int* __restrict__ slot_src, int* __restrict__ slot_dst,
                          int* __restrict__ slot_rel, int E) {
  int e = blockIdx.x * 256 + threadIdx.x;
  if (e < E) {
    int r   = etype[e];
    int pos = atomicAdd(&wcur[r], 1);
    slot_src[pos] = esrc[e];
    slot_dst[pos] = edst[e];
    slot_rel[pos] = r;
  }
}

// ------------------------- precision conversion ----------------------------

__global__ void k_cvt_x(const float* __restrict__ x, bf16_t* __restrict__ xb, int total) {
  int i = blockIdx.x * 256 + threadIdx.x;
  if (i < total) xb[i] = f32_to_bf16(x[i]);
}

// Pack all L*R relation matrices + L root matrices (each 256x256, row-major
// [K][N]) into B-fragment-major bf16: for (ntile, kb, lane, h) the element is
// B[kb*32 + (lane<16?0:16) + h][ntile*16 + lane%16]  -> one contiguous 32B
// fragment per lane at GEMM time.
__global__ void k_pack_w(const float* __restrict__ Wrel, const float* __restrict__ Wroot,
                         bf16_t* __restrict__ wrel_pk, bf16_t* __restrict__ wroot_pk) {
  int p = blockIdx.x * 256 + threadIdx.x;        // 0 .. 27*65536
  int m = p >> 16;                                // matrix id
  int q = p & 65535;
  int ntile = q >> 12;
  int kb    = (q >> 9) & 7;
  int lane  = (q >> 4) & 31;
  int h     = q & 15;
  int col = ntile * 16 + (lane & 15);
  int K   = kb * 32 + ((lane & 16) ? 16 : 0) + h;
  if (m < L_LAY * R_REL) {
    wrel_pk[(size_t)m * 65536 + q]  = f32_to_bf16(Wrel [(size_t)m * 65536 + K * D_DIM + col]);
  } else {
    int mr = m - L_LAY * R_REL;
    wroot_pk[(size_t)mr * 65536 + q] = f32_to_bf16(Wroot[(size_t)mr * 65536 + K * D_DIM + col]);
  }
}

// ------------------------- WMMA helpers ------------------------------------

// A fragment (16x32 bf16) from a row-major LDS tile, contiguous K layout:
// lanes 0-15: row=lane, K_local {0..15}; lanes 16-31: row=lane-16,
// K_local {16..31}.
__device__ __forceinline__ v16bf load_a_frag(const bf16_t* __restrict__ row_base, int kb, int lane) {
  const bf16_t* ar = row_base + kb * 32 + ((lane & 16) ? 16 : 0);
  v16bf a;
#pragma unroll
  for (int h = 0; h < 8; ++h) { a[h] = ar[h]; a[h + 8] = ar[h + 8]; }
  return a;
}

// ------------------------- root GEMM: agg = x@Wroot + bias -----------------

__global__ void __launch_bounds__(256)
k_root_gemm(const bf16_t* __restrict__ xb, const bf16_t* __restrict__ wpk,
            const float* __restrict__ bias, float* __restrict__ agg, int n_nodes) {
  __shared__ bf16_t At[16][D_DIM];
  const int tile = blockIdx.x;
  const int t    = threadIdx.x;
  {
    int row = t >> 4, seg = t & 15;
    int gr  = tile * 16 + row;
    uint4 v0 = {0, 0, 0, 0}, v1 = {0, 0, 0, 0};
    if (gr < n_nodes) {
      const uint4* p = (const uint4*)(xb + (size_t)gr * D_DIM + seg * 16);
      v0 = p[0]; v1 = p[1];
    }
    *(uint4*)&At[row][seg * 16]     = v0;
    *(uint4*)&At[row][seg * 16 + 8] = v1;
  }
  __syncthreads();

  const int lane = t & 31, wave = t >> 5;
  const int nt0 = wave, nt1 = wave + 8;
  const int mrow = lane & 15;
  v8f acc0 = {0, 0, 0, 0, 0, 0, 0, 0};
  v8f acc1 = {0, 0, 0, 0, 0, 0, 0, 0};

#pragma unroll
  for (int kb = 0; kb < 8; ++kb) {
    v16bf a = load_a_frag(&At[mrow][0], kb, lane);
    const v16bf b0 = *(const v16bf*)(wpk + (((nt0 * 8 + kb) * 32 + lane) << 4));
    const v16bf b1 = *(const v16bf*)(wpk + (((nt1 * 8 + kb) * 32 + lane) << 4));
    if (kb < 7) __builtin_prefetch(wpk + (((nt0 * 8 + kb + 1) * 32 + lane) << 4), 0, 1);
    acc0 = __builtin_amdgcn_wmma_f32_16x16x32_bf16(false, a, false, b0, (short)0, acc0, false, false);
    acc1 = __builtin_amdgcn_wmma_f32_16x16x32_bf16(false, a, false, b1, (short)0, acc1, false, false);
  }

  const int col0 = nt0 * 16 + (lane & 15);
  const int col1 = nt1 * 16 + (lane & 15);
  const int mbase = (lane >> 4) * 8;
  const float b0 = bias[col0], b1 = bias[col1];
#pragma unroll
  for (int v = 0; v < 8; ++v) {
    int row = tile * 16 + mbase + v;
    if (row < n_nodes) {
      agg[(size_t)row * D_DIM + col0] = acc0[v] + b0;
      agg[(size_t)row * D_DIM + col1] = acc1[v] + b1;
    }
  }
}

// ------------------- edge GEMM: agg[dst] += x[src] @ Wrel[rt] --------------

__global__ void __launch_bounds__(256)
k_edge_gemm(const bf16_t* __restrict__ xb, const bf16_t* __restrict__ wrel_pk,
            const int* __restrict__ slot_src, const int* __restrict__ slot_dst,
            const int* __restrict__ slot_rel, float* __restrict__ agg) {
  __shared__ bf16_t At[16][D_DIM];
  __shared__ int    DstS[16];
  __shared__ int    RelS;

  const int tile = blockIdx.x;
  const int t    = threadIdx.x;

  if (t == 0) RelS = (slot_src[tile * 16] >= 0) ? slot_rel[tile * 16] : -1;
  if (t < 16) DstS[t] = (slot_src[tile * 16 + t] >= 0) ? slot_dst[tile * 16 + t] : -1;
  {
    int row = t >> 4, seg = t & 15;
    int s = slot_src[tile * 16 + row];
    uint4 v0 = {0, 0, 0, 0}, v1 = {0, 0, 0, 0};
    if (s >= 0) {
      const uint4* p = (const uint4*)(xb + (size_t)s * D_DIM + seg * 16);
      v0 = p[0]; v1 = p[1];
    }
    *(uint4*)&At[row][seg * 16]     = v0;
    *(uint4*)&At[row][seg * 16 + 8] = v1;
  }
  __syncthreads();
  if (RelS < 0) return;                           // pure-padding tile (uniform)

  const bf16_t* wpk = wrel_pk + (size_t)RelS * 65536;
  const int lane = t & 31, wave = t >> 5;
  const int nt0 = wave, nt1 = wave + 8;
  const int mrow = lane & 15;
  v8f acc0 = {0, 0, 0, 0, 0, 0, 0, 0};
  v8f acc1 = {0, 0, 0, 0, 0, 0, 0, 0};

#pragma unroll
  for (int kb = 0; kb < 8; ++kb) {
    v16bf a = load_a_frag(&At[mrow][0], kb, lane);
    const v16bf b0 = *(const v16bf*)(wpk + (((nt0 * 8 + kb) * 32 + lane) << 4));
    const v16bf b1 = *(const v16bf*)(wpk + (((nt1 * 8 + kb) * 32 + lane) << 4));
    if (kb < 7) __builtin_prefetch(wpk + (((nt0 * 8 + kb + 1) * 32 + lane) << 4), 0, 1);
    acc0 = __builtin_amdgcn_wmma_f32_16x16x32_bf16(false, a, false, b0, (short)0, acc0, false, false);
    acc1 = __builtin_amdgcn_wmma_f32_16x16x32_bf16(false, a, false, b1, (short)0, acc1, false, false);
  }

  const int col0 = nt0 * 16 + (lane & 15);
  const int col1 = nt1 * 16 + (lane & 15);
  const int mbase = (lane >> 4) * 8;
#pragma unroll
  for (int v = 0; v < 8; ++v) {
    int d = DstS[mbase + v];
    if (d >= 0) {
      atomic_add_f32(&agg[(size_t)d * D_DIM + col0], acc0[v]);
      atomic_add_f32(&agg[(size_t)d * D_DIM + col1], acc1[v]);
    }
  }
}

// ------------------------- ReLU + requantize -------------------------------

__global__ void k_relu(const float* __restrict__ agg, bf16_t* __restrict__ xb,
                       float* __restrict__ out, int total, int write_out) {
  int i = blockIdx.x * 256 + threadIdx.x;
  if (i < total) {
    float v = agg[i];
    v = v > 0.0f ? v : 0.0f;
    xb[i] = f32_to_bf16(v);
    if (write_out) out[i] = v;
  }
}

// ------------------------- host launch -------------------------------------

extern "C" void kernel_launch(void* const* d_in, const int* in_sizes, int n_in,
                              void* d_out, int out_size, void* d_ws, size_t ws_size,
                              hipStream_t stream) {
  const float* x      = (const float*)d_in[0];
  const int*   eidx   = (const int*)  d_in[1];   // [2,E] flattened: src then dst
  const int*   etype  = (const int*)  d_in[2];
  const float* Wrel   = (const float*)d_in[3];   // [L,R,256,256]
  const float* Wroot  = (const float*)d_in[4];   // [L,256,256]
  const float* bias   = (const float*)d_in[5];   // [L,256]

  const int N = in_sizes[0] / D_DIM;             // 50000
  const int E = in_sizes[2];                     // 250000
  const int total = N * D_DIM;
  const int tiles = (E + 15) / 16 + R_REL;       // covers 16-aligned bucket padding
  const int slots = tiles * 16;

  // ---- carve workspace ----
  char* ws = (char*)d_ws;
  size_t off = 0;
  auto carve = [&](size_t bytes) { char* p = ws + off; off = (off + bytes + 255) & ~(size_t)255; return p; };
  bf16_t* xb       = (bf16_t*)carve((size_t)total * sizeof(bf16_t));
  float*  agg      = (float*) carve((size_t)total * sizeof(float));
  bf16_t* wrel_pk  = (bf16_t*)carve((size_t)L_LAY * R_REL * 65536 * sizeof(bf16_t));
  bf16_t* wroot_pk = (bf16_t*)carve((size_t)L_LAY * 65536 * sizeof(bf16_t));
  int*    cnt      = (int*)   carve(R_REL * sizeof(int));
  int*    wcur     = (int*)   carve(R_REL * sizeof(int));
  int*    slot_src = (int*)   carve((size_t)slots * sizeof(int));
  int*    slot_dst = (int*)   carve((size_t)slots * sizeof(int));
  int*    slot_rel = (int*)   carve((size_t)slots * sizeof(int));
  (void)ws_size; (void)n_in; (void)out_size;

  const int gE = (E + 255) / 256;
  const int gS = (slots + 255) / 256;
  const int gT = (total + 255) / 256;

  // ---- bucketing (relation-sorted, 16-aligned; reused by all layers) ----
  k_zero_counts<<<1, 32, 0, stream>>>(cnt);
  k_count<<<gE, 256, 0, stream>>>(etype, cnt, E);
  k_scan<<<1, 1, 0, stream>>>(cnt, wcur);
  k_fill<<<gS, 256, 0, stream>>>(slot_src, slots);
  k_scatter<<<gE, 256, 0, stream>>>(eidx, eidx + E, etype, wcur,
                                    slot_src, slot_dst, slot_rel, E);

  // ---- precision prep ----
  k_cvt_x<<<gT, 256, 0, stream>>>(x, xb, total);
  k_pack_w<<<(L_LAY * (R_REL + 1) * 65536) / 256, 256, 0, stream>>>(Wrel, Wroot, wrel_pk, wroot_pk);

  // ---- layers ----
  const int node_tiles = (N + 15) / 16;
  for (int l = 0; l < L_LAY; ++l) {
    k_root_gemm<<<node_tiles, 256, 0, stream>>>(xb, wroot_pk + (size_t)l * 65536,
                                                bias + (size_t)l * D_DIM, agg, N);
    k_edge_gemm<<<tiles, 256, 0, stream>>>(xb, wrel_pk + (size_t)l * R_REL * 65536,
                                           slot_src, slot_dst, slot_rel, agg);
    k_relu<<<gT, 256, 0, stream>>>(agg, xb, (float*)d_out, total, l == L_LAY - 1 ? 1 : 0);
  }
}
